// MultiHeadAttention_80942953660780
// MI455X (gfx1250) — matrix operand order repaired and checked by reading, not compiled, not run on hardware
//
#include <hip/hip_runtime.h>
#include <hip/hip_bf16.h>

// ---------------------------------------------------------------------------
// MHA forward for MI455X (gfx1250).
// * All matmuls on V_WMMA_F32_16X16X32_BF16 (operands pre-converted to bf16
//   once, so GEMM K-loops are pure global_load_b128 + v_wmma).
// * Flash-attention online softmax, wave32 half-wave shfl reductions.
// * V tiles DMA'd into LDS by the Tensor Data Mover (tensor_load_to_lds),
//   double-buffered and synchronized with s_wait_tensorcnt.
//   B=2, S=2048, D=1024, H=16, hd=64, scale = 1/hd (folded into Q).
// ---------------------------------------------------------------------------

typedef __attribute__((ext_vector_type(16))) __bf16        v16bf;
typedef __attribute__((ext_vector_type(8)))  float         v8f;
typedef __attribute__((ext_vector_type(4)))  unsigned int  u32x4;
typedef __attribute__((ext_vector_type(4)))  float         f32x4;
typedef __attribute__((ext_vector_type(8)))  int           i32x8;
typedef __attribute__((ext_vector_type(4)))  int           i32x4;

#define BATCH 2
#define SEQ   2048
#define DIM   1024
#define HEADS 16
#define HD    64

union AF {
  v16bf v;
  unsigned short u[16];
  u32x4 q[2];
};

static __device__ __forceinline__ unsigned short f2bf(float f) {
  unsigned int u = __builtin_bit_cast(unsigned int, f);
  unsigned int r = u + 0x7FFFu + ((u >> 16) & 1u);   // round-to-nearest-even
  return (unsigned short)(r >> 16);
}

static __device__ __forceinline__ unsigned int pack2bf(float lo, float hi) {
  return (unsigned int)f2bf(lo) | ((unsigned int)f2bf(hi) << 16);
}

static __device__ __forceinline__ v8f wmma_bf16(v16bf a, v16bf b, v8f c) {
  return __builtin_amdgcn_wmma_f32_16x16x32_bf16(
      /*neg_a=*/false, a, /*neg_b=*/false, b,
      /*c_mod=*/(short)0, c, /*reuse_a=*/false, /*reuse_b=*/false);
}

// A fragment (16x32, M x K): m = lane&15; k = kb + e, kb = (lane>=16)?16:0.
static __device__ __forceinline__ v16bf load_a_bf16(const unsigned short* rowptr, int lane) {
  const int kb = (lane & 16) ? 16 : 0;
  AF f;
  f.q[0] = *(const u32x4*)(rowptr + kb);
  f.q[1] = *(const u32x4*)(rowptr + kb + 8);
  return f.v;
}

// B fragment (32x16, K x N): n = lane&15; k = kb + e, kb = (lane>=16)?16:0.
static __device__ __forceinline__ v16bf load_b_bf16(const unsigned short* rowptr, int lane) {
  const int kb = (lane & 16) ? 16 : 0;
  AF f;
  f.q[0] = *(const u32x4*)(rowptr + kb);
  f.q[1] = *(const u32x4*)(rowptr + kb + 8);
  return f.v;
}

// Tensor Data Mover: DMA a 32-row x 64-col bf16 tile (row stride 64 elem)
// from global memory into LDS.  D# packed per CDNA5 ISA 08_async_tensor §8.
// This toolchain exposes the 6-arg builtin form.
static __device__ __forceinline__ void tdm_load_v_tile(unsigned lds_off,
                                                       const unsigned short* gptr) {
  const unsigned long long ga = (unsigned long long)(uintptr_t)gptr;
  u32x4 g0;
  g0[0] = 1u;                                             // count=1, user mode
  g0[1] = lds_off;                                        // lds_addr (bytes)
  g0[2] = (unsigned)(ga & 0xFFFFFFFFu);                   // global_addr[31:0]
  g0[3] = (unsigned)((ga >> 32) & 0x01FFFFFFu)            // global_addr[56:32]
          | 0x80000000u;                                  // type = 2 (image)
  i32x8 g1;
  g1[0] = 0x00010000;          // wg_mask=0, data_size=1 (2 bytes)
  g1[1] = (int)(64u << 16);    // tensor_dim0 = 64 elements / row
  g1[2] = (int)(2048u << 16);  // tensor_dim1 = 2048 rows
  g1[3] = (int)(64u << 16);    // tile_dim0 = 64
  g1[4] = 32;                  // tile_dim1 = 32, tile_dim2 = 0
  g1[5] = 64;                  // tensor_dim0_stride = 64 elements
  g1[6] = 0;
  g1[7] = 0;
  const i32x4 gz4 = {0, 0, 0, 0};                         // 2-D tensor
  const i32x8 gz8 = {0, 0, 0, 0, 0, 0, 0, 0};
  __builtin_amdgcn_tensor_load_to_lds(g0, g1, gz4, gz4, gz8, 0);
}

// ---------------------------------------------------------------------------
// Kernel 0: one-shot f32 -> bf16 conversion (8 elements / thread).
// ---------------------------------------------------------------------------
__global__ __launch_bounds__(256) void cvt_bf16_kernel(
    const float* __restrict__ src, unsigned short* __restrict__ dst, int n8) {
  const int i = blockIdx.x * blockDim.x + threadIdx.x;
  if (i >= n8) return;
  const f32x4 a = *(const f32x4*)(src + (size_t)i * 8);
  const f32x4 b = *(const f32x4*)(src + (size_t)i * 8 + 4);
  u32x4 o;
  o[0] = pack2bf(a[0], a[1]);
  o[1] = pack2bf(a[2], a[3]);
  o[2] = pack2bf(b[0], b[1]);
  o[3] = pack2bf(b[2], b[3]);
  *(u32x4*)(dst + (size_t)i * 8) = o;
}

// ---------------------------------------------------------------------------
// Kernel 1: QKV projection.  M = 4096, N = 3072, K = 1024.
// Block = 256 (8 waves), wave tile = 16x128 (8 WMMA / A-fragment).
// Q results pre-scaled by 1/hd; scattered into [B,H,S,hd] bf16 buffers.
// ---------------------------------------------------------------------------
__global__ __launch_bounds__(256) void qkv_proj_kernel(
    const unsigned short* __restrict__ Xb, const unsigned short* __restrict__ Wb,
    unsigned short* __restrict__ Qb, unsigned short* __restrict__ Kb,
    unsigned short* __restrict__ Vb) {
  const int lane = threadIdx.x & 31;
  const int wave = threadIdx.x >> 5;
  const int m0 = blockIdx.y * 16;
  const int n0 = blockIdx.x * 1024 + wave * 128;

  v8f acc[8];
  const v8f vzero = {};
#pragma unroll
  for (int t = 0; t < 8; ++t) acc[t] = vzero;

  const unsigned short* arow = Xb + (size_t)(m0 + (lane & 15)) * DIM;

  for (int kk = 0; kk < DIM; kk += 32) {
    __builtin_prefetch(arow + kk + 256, 0, 0);
    const v16bf A = load_a_bf16(arow + kk, lane);
#pragma unroll
    for (int t = 0; t < 8; ++t) {
      const int n = n0 + t * 16 + (lane & 15);
      const v16bf Bm = load_b_bf16(Wb + (size_t)n * DIM + kk, lane);
      acc[t] = wmma_bf16(A, Bm, acc[t]);
    }
  }

#pragma unroll
  for (int t = 0; t < 8; ++t) {
    const int n = n0 + t * 16 + (lane & 15);
    const int h = n / 192;
    const int w = n % 192;
    const int sel = w >> 6;                              // 0=q 1=k 2=v
    const int hdi = w & 63;
    unsigned short* dst = (sel == 0) ? Qb : ((sel == 1) ? Kb : Vb);
    const float scl = (sel == 0) ? 0.015625f : 1.0f;     // fold 1/hd into Q
#pragma unroll
    for (int r = 0; r < 8; ++r) {
      const int m = m0 + r + ((lane >> 4) << 3);
      const int b = m >> 11;
      const int s = m & 2047;
      dst[(((size_t)(b * HEADS + h) * SEQ + s) << 6) + hdi] = f2bf(acc[t][r] * scl);
    }
  }
}

// ---------------------------------------------------------------------------
// Kernel 2: flash attention.  One wave per (b, h, 16-query tile).
// Per 32-key block: TDM prefetch of next V tile into LDS (double buffered),
// 4 WMMA scores, online softmax, P via LDS re-layout, 4 WMMA for O += P*V.
// ---------------------------------------------------------------------------
__global__ __launch_bounds__(32) void attn_kernel(
    const unsigned short* __restrict__ Qb, const unsigned short* __restrict__ Kb,
    const unsigned short* __restrict__ Vb, unsigned short* __restrict__ Ob) {
  __shared__ __align__(64) unsigned short Plds[16][32];
  __shared__ __align__(128) unsigned short Vlds[2][32][64];

  const int lane = threadIdx.x;
  const int qt = blockIdx.x & 127;
  const int h  = (blockIdx.x >> 7) & 15;
  const int b  = blockIdx.x >> 11;
  const int bh = b * HEADS + h;

  const unsigned short* Qbase = Qb + (size_t)bh * SEQ * HD;
  const unsigned short* Kbase = Kb + (size_t)bh * SEQ * HD;
  const unsigned short* Vbase = Vb + (size_t)bh * SEQ * HD;

  // LDS byte offsets of the two V buffers (LDS aperture: addr[31:0] = offset).
  const unsigned voff0 = (unsigned)(uintptr_t)&Vlds[0][0][0];
  const unsigned voff1 = (unsigned)(uintptr_t)&Vlds[1][0][0];

  const int mq = qt * 16 + (lane & 15);
  const v16bf aQ0 = load_a_bf16(Qbase + (size_t)mq * HD, lane);
  const v16bf aQ1 = load_a_bf16(Qbase + (size_t)mq * HD + 32, lane);

  float Mr[8], Lr[8];
  v8f o[4];
  const v8f vzero = {};
#pragma unroll
  for (int r = 0; r < 8; ++r) { Mr[r] = -1e30f; Lr[r] = 0.0f; }
#pragma unroll
  for (int c = 0; c < 4; ++c) o[c] = vzero;

  const int mMax = qt * 16 + 15;
  const int mRowBase = qt * 16 + ((lane >> 4) << 3);

  // Prime the pipeline: DMA first V tile (keys 0..31) into buffer 0.
  tdm_load_v_tile(voff0, Vbase);
  int cur = 0;

  for (int j0 = 0; j0 <= mMax; j0 += 32) {
    const bool hasNext = (j0 + 32) <= mMax;
    // Issue DMA for the next V tile while we compute on the current one.
    if (hasNext)
      tdm_load_v_tile(cur ? voff0 : voff1, Vbase + (size_t)(j0 + 32) * HD);

    // ----- scores: two 16x16 tiles over d-chunks (scale pre-folded) -----
    v8f sc[2];
#pragma unroll
    for (int t = 0; t < 2; ++t) {
      sc[t] = vzero;
      const int nk = j0 + t * 16 + (lane & 15);
      const unsigned short* krow = Kbase + (size_t)nk * HD;
      const v16bf bK0 = load_b_bf16(krow, lane);
      const v16bf bK1 = load_b_bf16(krow + 32, lane);
      sc[t] = wmma_bf16(aQ0, bK0, sc[t]);
      sc[t] = wmma_bf16(aQ1, bK1, sc[t]);
    }

    // ----- causal mask + online softmax update -----
    const int k0 = j0 + (lane & 15);
#pragma unroll
    for (int r = 0; r < 8; ++r) {
      const int m = mRowBase + r;
      float s0 = sc[0][r];
      float s1 = sc[1][r];
      if (k0 > m)      s0 = -1e30f;
      if (k0 + 16 > m) s1 = -1e30f;
      float mx = fmaxf(s0, s1);
      mx = fmaxf(mx, __shfl_xor(mx, 1));
      mx = fmaxf(mx, __shfl_xor(mx, 2));
      mx = fmaxf(mx, __shfl_xor(mx, 4));
      mx = fmaxf(mx, __shfl_xor(mx, 8));
      const float Mn = fmaxf(Mr[r], mx);
      const float alpha = __expf(Mr[r] - Mn);
      const float p0 = __expf(s0 - Mn);
      const float p1 = __expf(s1 - Mn);
      float rs = p0 + p1;
      rs += __shfl_xor(rs, 1);
      rs += __shfl_xor(rs, 2);
      rs += __shfl_xor(rs, 4);
      rs += __shfl_xor(rs, 8);
      Lr[r] = Lr[r] * alpha + rs;
      Mr[r] = Mn;
#pragma unroll
      for (int c = 0; c < 4; ++c) o[c][r] *= alpha;
      const int mr = r + ((lane >> 4) << 3);
      Plds[mr][lane & 15]        = f2bf(p0);
      Plds[mr][16 + (lane & 15)] = f2bf(p1);
    }
    __syncthreads();

    // ----- P (16x32) C-layout -> A-fragment via LDS -----
    const v16bf aP = load_a_bf16(&Plds[lane & 15][0], lane);
    __syncthreads();

    // Wait for the current V tile's DMA (allow the next one to stay in
    // flight: in-order TENSORcnt completion => <=1 outstanding is enough).
    if (hasNext) __builtin_amdgcn_s_wait_tensorcnt(1);
    else         __builtin_amdgcn_s_wait_tensorcnt(0);

    // ----- O += P * V from LDS (B-fragment transpose gather) -----
    const int kb = (lane & 16) ? 16 : 0;
#pragma unroll
    for (int c = 0; c < 4; ++c) {
      const int nhd = c * 16 + (lane & 15);
      AF f;
#pragma unroll
      for (int e = 0; e < 16; ++e) f.u[e] = Vlds[cur][kb + e][nhd];
      o[c] = wmma_bf16(aP, f.v, o[c]);
    }
    cur ^= 1;
  }

  // ----- finalize: divide by row sums, write O as bf16 [B,S,D] -----
  float inv[8];
#pragma unroll
  for (int r = 0; r < 8; ++r) inv[r] = 1.0f / Lr[r];
#pragma unroll
  for (int c = 0; c < 4; ++c) {
    const int col = h * HD + c * 16 + (lane & 15);
#pragma unroll
    for (int r = 0; r < 8; ++r) {
      const int m = qt * 16 + r + ((lane >> 4) << 3);
      Ob[((size_t)b * SEQ + m) * DIM + col] = f2bf(o[c][r] * inv[r]);
    }
  }
}

// ---------------------------------------------------------------------------
// Kernel 3: output projection.  M = 4096, N = 1024, K = 1024.
// Wave tile 16x128, all-bf16 operands, bias epilogue, f32 output.
// ---------------------------------------------------------------------------
__global__ __launch_bounds__(256) void out_proj_kernel(
    const unsigned short* __restrict__ Ob, const unsigned short* __restrict__ Wb,
    const float* __restrict__ bout, float* __restrict__ out) {
  const int lane = threadIdx.x & 31;
  const int wave = threadIdx.x >> 5;
  const int m0 = blockIdx.y * 16;
  const int n0 = blockIdx.x * 1024 + wave * 128;

  v8f acc[8];
  const v8f vzero = {};
#pragma unroll
  for (int t = 0; t < 8; ++t) acc[t] = vzero;

  const unsigned short* arow = Ob + (size_t)(m0 + (lane & 15)) * DIM;

  for (int kk = 0; kk < DIM; kk += 32) {
    __builtin_prefetch(arow + kk + 256, 0, 0);
    const v16bf A = load_a_bf16(arow + kk, lane);
#pragma unroll
    for (int t = 0; t < 8; ++t) {
      const int n = n0 + t * 16 + (lane & 15);
      const v16bf Bm = load_b_bf16(Wb + (size_t)n * DIM + kk, lane);
      acc[t] = wmma_bf16(A, Bm, acc[t]);
    }
  }

#pragma unroll
  for (int t = 0; t < 8; ++t) {
    const int n = n0 + t * 16 + (lane & 15);
    const float bias = bout[n];
#pragma unroll
    for (int r = 0; r < 8; ++r) {
      const int m = m0 + r + ((lane >> 4) << 3);
      out[(size_t)m * DIM + n] = acc[t][r] + bias;
    }
  }
}

// ---------------------------------------------------------------------------
extern "C" void kernel_launch(void* const* d_in, const int* in_sizes, int n_in,
                              void* d_out, int out_size, void* d_ws, size_t ws_size,
                              hipStream_t stream) {
  const float* x    = (const float*)d_in[0];
  // d_in[1] = causal mask (tril) -- applied analytically in attn_kernel.
  const float* wqkv = (const float*)d_in[2];
  const float* wout = (const float*)d_in[3];
  const float* bout = (const float*)d_in[4];
  float* out = (float*)d_out;

  // Workspace layout (all bf16 / ushort elements): ~48 MB total.
  const size_t per = (size_t)BATCH * HEADS * SEQ * HD;   // 4,194,304
  unsigned short* Qb    = (unsigned short*)d_ws;
  unsigned short* Kb    = Qb + per;
  unsigned short* Vb    = Kb + per;
  unsigned short* Ob    = Vb + per;
  unsigned short* Xb    = Ob + per;
  unsigned short* Wqkvb = Xb + per;
  unsigned short* Woutb = Wqkvb + (size_t)3 * DIM * DIM;

  // 0) one-shot bf16 conversions (memory-bound, ~30 MB of traffic)
  cvt_bf16_kernel<<<dim3(2048), 256, 0, stream>>>(x,    Xb,    524288);
  cvt_bf16_kernel<<<dim3(1536), 256, 0, stream>>>(wqkv, Wqkvb, 393216);
  cvt_bf16_kernel<<<dim3(512),  256, 0, stream>>>(wout, Woutb, 131072);

  // 1) QKV projection: grid (3072/1024, 4096/16)
  qkv_proj_kernel<<<dim3(3, 256), 256, 0, stream>>>(Xb, Wqkvb, Qb, Kb, Vb);

  // 2) flash attention: one wave per (b,h,16-row query tile)
  attn_kernel<<<dim3(BATCH * HEADS * (SEQ / 16)), 32, 0, stream>>>(Qb, Kb, Vb, Ob);

  // 3) output projection + bias: grid (1024/1024, 4096/16)
  out_proj_kernel<<<dim3(1, 256), 256, 0, stream>>>(Ob, Woutb, bout, out);
}
